// GATConv_42786464203342
// MI455X (gfx1250) — matrix operand order repaired and checked by reading, not compiled, not run on hardware
//
#include <hip/hip_runtime.h>
#include <math.h>

// ---------------- problem constants ----------------
#define NODES 50000
#define CC    256
#define HH    8
#define HDIM  32
#define EE    1600000
#define FF    1024          // 4*C

typedef __attribute__((ext_vector_type(16))) __bf16 v16bf;
typedef __attribute__((ext_vector_type(8)))  float  v8f;

// ---------------- helpers ----------------
__device__ __forceinline__ unsigned short f2bf(float f) {
  union { float f; unsigned int u; } c; c.f = f;
  unsigned int u = c.u;
  u += 0x7FFFu + ((u >> 16) & 1u);           // round-to-nearest-even
  return (unsigned short)(u >> 16);
}
// order-preserving float<->uint map (for atomicMax-based segment max)
__device__ __forceinline__ unsigned int ordEnc(float f) {
  union { float f; unsigned int u; } c; c.f = f;
  return (c.u & 0x80000000u) ? ~c.u : (c.u | 0x80000000u);
}
__device__ __forceinline__ float ordDec(unsigned int e) {
  union { unsigned int u; float f; } c;
  c.u = (e & 0x80000000u) ? (e & 0x7FFFFFFFu) : ~e;
  return c.f;
}

// ---------------- elementwise utility kernels ----------------
__global__ void cvt_f32_bf16_kernel(const float* __restrict__ in,
                                    unsigned short* __restrict__ out, size_t n) {
  size_t i = (size_t)blockIdx.x * blockDim.x + threadIdx.x;
  if (i < n) out[i] = f2bf(in[i]);
}
__global__ void fill_f32_kernel(float* __restrict__ p, float v, size_t n) {
  size_t i = (size_t)blockIdx.x * blockDim.x + threadIdx.x;
  if (i < n) p[i] = v;
}
__global__ void fill_u32_kernel(unsigned int* __restrict__ p, unsigned int v, size_t n) {
  size_t i = (size_t)blockIdx.x * blockDim.x + threadIdx.x;
  if (i < n) p[i] = v;
}

// ---------------- bf16 WMMA GEMM: out[M,N] = A[M,K] * W[N,K]^T + bias ----------------
// Block = 8 waves sharing one 64-wide N strip of W, staged in LDS in 256-K chunks
// (cooperative fill, +8 element row pad to spread lanes across LDS banks).
// Each wave computes a 32x64 strip: 2 M-subtiles x 4 N-tiles. All 4 B fragments
// are preloaded from LDS before the 8-WMMA burst so WMMAs issue back-to-back.
// A-frag layout (16-bit 16x32): lane half hi=lane/16, M=lane%16;
//   vgpr pairs 0..3 -> K = 8*hi + {0..7}, pairs 4..7 -> K = 16 + 8*hi + {0..7}
// B-frag layout (16-bit 32x16): N=lane%16; same K order as A-frag
#define KB_CHUNK 256
#define KPAD (KB_CHUNK + 8)
__global__ __launch_bounds__(256) void gemm_bf16_wmma_kernel(
    const unsigned short* __restrict__ A, const unsigned short* __restrict__ W,
    const float* __restrict__ bias, const float* __restrict__ resid,
    float* __restrict__ outF, unsigned short* __restrict__ outB,
    int M, int N, int K, int act /*0=none,1=exact gelu*/) {
  __shared__ __align__(16) unsigned short ldsB[64 * KPAD];

  const int lane = threadIdx.x & 31;
  const int wvid = threadIdx.x >> 5;
  const int mTiles  = M >> 4;
  const int mBlocks = (mTiles + 15) >> 4;        // 16 m-tiles per block (2 per wave)
  const int mb = blockIdx.x % mBlocks;
  const int ng = blockIdx.x / mBlocks;
  const int t0 = mb * 16 + wvid * 2;             // first m-tile of this wave
  const int t1 = t0 + 1;
  const bool valid0 = (t0 < mTiles);
  const bool valid1 = (t1 < mTiles);
  const int t0c = valid0 ? t0 : (mTiles - 1);    // clamped for safe loads
  const int t1c = valid1 ? t1 : (mTiles - 1);
  const int nBase = ng << 6;
  const int hi  = lane >> 4;
  const int l16 = lane & 15;

  v8f acc[8];
#pragma unroll
  for (int t = 0; t < 8; ++t) acc[t] = (v8f){0,0,0,0,0,0,0,0};

  const unsigned short* arow0 = A + (size_t)((t0c << 4) + l16) * K;
  const unsigned short* arow1 = A + (size_t)((t1c << 4) + l16) * K;

  for (int kc = 0; kc < K; kc += KB_CHUNK) {
    __syncthreads();
    // cooperative LDS fill of W[nBase..nBase+63, kc..kc+255]
    {
      const int r   = threadIdx.x >> 2;   // 0..63
      const int qtr = threadIdx.x & 3;    // 0..3
      const unsigned short* wsrc = W + (size_t)(nBase + r) * K + kc + qtr * 64;
      unsigned short* ldst = ldsB + r * KPAD + qtr * 64;
#pragma unroll
      for (int i = 0; i < 8; ++i)
        *(uint4*)(ldst + i * 8) = *(const uint4*)(wsrc + i * 8);
    }
    __syncthreads();
    if (kc + KB_CHUNK < K) {
      __builtin_prefetch(arow0 + kc + KB_CHUNK, 0, 1);   // global_prefetch next A chunk
      __builtin_prefetch(arow1 + kc + KB_CHUNK, 0, 1);
    }
    for (int kb = 0; kb < KB_CHUNK; kb += 32) {
      union { unsigned int u[8]; v16bf v; } af0, af1;
#pragma unroll
      for (int j = 0; j < 4; ++j) {
        af0.u[j]   = *(const unsigned int*)(arow0 + kc + kb + 8*hi + 2*j);
        af0.u[4+j] = *(const unsigned int*)(arow0 + kc + kb + 16 + 8*hi + 2*j);
        af1.u[j]   = *(const unsigned int*)(arow1 + kc + kb + 8*hi + 2*j);
        af1.u[4+j] = *(const unsigned int*)(arow1 + kc + kb + 16 + 8*hi + 2*j);
      }
      union { unsigned int u[8]; v16bf v; } bfr[4];
#pragma unroll
      for (int t = 0; t < 4; ++t) {
        const unsigned short* brow = ldsB + (t*16 + l16) * KPAD + kb + 8*hi;
#pragma unroll
        for (int j = 0; j < 8; ++j)
          bfr[t].u[j] = *(const unsigned int*)(brow + 2*j + (j >= 4 ? 8 : 0));
      }
#pragma unroll
      for (int t = 0; t < 4; ++t) {
        acc[t]   = __builtin_amdgcn_wmma_f32_16x16x32_bf16(
            false, af0.v, false, bfr[t].v, (short)0, acc[t],   false, false);
        acc[4+t] = __builtin_amdgcn_wmma_f32_16x16x32_bf16(
            false, af1.v, false, bfr[t].v, (short)0, acc[4+t], false, false);
      }
    }
  }
  // epilogue: bias (+residual) (+gelu), write f32 and/or bf16
#pragma unroll
  for (int s = 0; s < 2; ++s) {
    if (s == 0 ? !valid0 : !valid1) continue;
    const int mBase = (s == 0 ? t0 : t1) << 4;
#pragma unroll
    for (int t = 0; t < 4; ++t) {
      const int n  = nBase + t*16 + l16;
      const float bn = bias ? bias[n] : 0.0f;
      const v8f a = acc[s*4 + t];
#pragma unroll
      for (int r = 0; r < 8; ++r) {
        const int m = mBase + r + 8*hi;
        const size_t idx = (size_t)m * N + n;
        float v = a[r] + bn;
        if (resid) v += resid[idx];
        if (act) v = 0.5f * v * (1.0f + erff(v * 0.70710678118654752f));
        if (outF) outF[idx] = v;
        if (outB) outB[idx] = f2bf(v);
      }
    }
  }
}

// ---------------- edge pass 1: score + segment max (wave per edge) ----------------
__global__ __launch_bounds__(256) void edge_score_kernel(
    const int* __restrict__ esrc, const int* __restrict__ edst,
    const float* __restrict__ q, const float* __restrict__ k,
    float* __restrict__ score, unsigned int* __restrict__ smaxe) {
  const int lane = threadIdx.x & 31;
  const int e = blockIdx.x * 8 + (threadIdx.x >> 5);
  if (e >= EE) return;
  const int s = esrc[e], d = edst[e];
  const float* qd = q + (size_t)d * CC;
  const float* ks = k + (size_t)s * CC;
  float p[HH];
#pragma unroll
  for (int j = 0; j < HH; ++j)
    p[j] = qd[j*HDIM + lane] * ks[j*HDIM + lane];
#pragma unroll
  for (int j = 0; j < HH; ++j)
#pragma unroll
    for (int off = 16; off > 0; off >>= 1)
      p[j] += __shfl_xor(p[j], off, 32);
  if (lane < HH) {
    float sc = p[lane] * 0.17677669529663687f;   // 1/sqrt(32)
    sc = fminf(10.0f, fmaxf(-10.0f, sc));
    score[(size_t)e*HH + lane] = sc;
    atomicMax(&smaxe[(size_t)d*HH + lane], ordEnc(sc));
  }
}

// ---------------- edge pass 2: exp + denominator ----------------
__global__ void edge_exp_kernel(const int* __restrict__ edst,
                                float* __restrict__ score,
                                const unsigned int* __restrict__ smaxe,
                                float* __restrict__ den) {
  size_t i = (size_t)blockIdx.x * blockDim.x + threadIdx.x;
  if (i >= (size_t)EE * HH) return;
  const int e = (int)(i >> 3);
  const int h = (int)(i & 7);
  const int d = edst[e];
  float ex = expf(score[i] - ordDec(smaxe[(size_t)d*HH + h]));
  score[i] = ex;
  atomicAdd(&den[(size_t)d*HH + h], ex);
}

// ---------------- edge pass 3: alpha * v scatter-add (wave per edge) ----------------
__global__ __launch_bounds__(256) void edge_aggr_kernel(
    const int* __restrict__ esrc, const int* __restrict__ edst,
    const float* __restrict__ score, const float* __restrict__ den,
    const float* __restrict__ v, float* __restrict__ wvout) {
  const int lane = threadIdx.x & 31;
  const int e = blockIdx.x * 8 + (threadIdx.x >> 5);
  if (e >= EE) return;
  const int s = esrc[e], d = edst[e];
  float a = 0.0f;
  if (lane < HH) a = score[(size_t)e*HH + lane] / den[(size_t)d*HH + lane];
  const float* vs = v + (size_t)s * CC;
  float* wd = wvout + (size_t)d * CC;
#pragma unroll
  for (int j = 0; j < HH; ++j) {
    const float aj = __shfl(a, j, 32);
    atomicAdd(&wd[j*HDIM + lane], aj * vs[j*HDIM + lane]);
  }
}

// ---------------- layernorm (wave per row) -> bf16 ----------------
__global__ __launch_bounds__(256) void ln_kernel(
    const float* __restrict__ x, const float* __restrict__ g,
    const float* __restrict__ b, unsigned short* __restrict__ out) {
  const int lane = threadIdx.x & 31;
  const int row = blockIdx.x * 8 + (threadIdx.x >> 5);
  if (row >= NODES) return;
  const float* xr = x + (size_t)row * CC;
  float v[8]; float s = 0.0f;
#pragma unroll
  for (int j = 0; j < 8; ++j) { v[j] = xr[j*32 + lane]; s += v[j]; }
#pragma unroll
  for (int off = 16; off > 0; off >>= 1) s += __shfl_xor(s, off, 32);
  const float mu = s * (1.0f/256.0f);
  float var = 0.0f;
#pragma unroll
  for (int j = 0; j < 8; ++j) { float dd = v[j] - mu; var += dd*dd; }
#pragma unroll
  for (int off = 16; off > 0; off >>= 1) var += __shfl_xor(var, off, 32);
  const float rs = rsqrtf(var * (1.0f/256.0f) + 1e-5f);
#pragma unroll
  for (int j = 0; j < 8; ++j) {
    const int c = j*32 + lane;
    out[(size_t)row*CC + c] = f2bf((v[j] - mu) * rs * g[c] + b[c]);
  }
}

// ---------------- host side ----------------
static inline char* carve(char*& p, size_t bytes) {
  char* r = p;
  p += (bytes + 255) & ~(size_t)255;
  return r;
}
static inline int ceil_div(size_t a, size_t b) { return (int)((a + b - 1) / b); }

static inline void launch_gemm(const unsigned short* A, const unsigned short* W,
                               const float* bias, const float* resid,
                               float* outF, unsigned short* outB,
                               int M, int N, int K, int act, hipStream_t stream) {
  int mBlocks = ((M >> 4) + 15) >> 4;
  int blocks = mBlocks * (N >> 6);
  gemm_bf16_wmma_kernel<<<blocks, dim3(256), 0, stream>>>(A, W, bias, resid,
                                                          outF, outB, M, N, K, act);
}

extern "C" void kernel_launch(void* const* d_in, const int* in_sizes, int n_in,
                              void* d_out, int out_size, void* d_ws, size_t ws_size,
                              hipStream_t stream) {
  (void)in_sizes; (void)n_in; (void)out_size; (void)ws_size;
  const float* feat = (const float*)d_in[0];
  const int*   esrc = (const int*)  d_in[1];
  const int*   edst = (const int*)  d_in[2];
  const float* Wq = (const float*)d_in[3];  const float* bq = (const float*)d_in[4];
  const float* Wk = (const float*)d_in[5];  const float* bk = (const float*)d_in[6];
  const float* Wv = (const float*)d_in[7];  const float* bv = (const float*)d_in[8];
  const float* Wp = (const float*)d_in[9];  const float* bp = (const float*)d_in[10];
  const float* lng = (const float*)d_in[11]; const float* lnb = (const float*)d_in[12];
  const float* W1 = (const float*)d_in[13]; const float* b1 = (const float*)d_in[14];
  const float* W2 = (const float*)d_in[15]; const float* b2 = (const float*)d_in[16];
  float* out = (float*)d_out;

  // ---- workspace carve (~340 MB) ----
  char* p = (char*)d_ws;
  unsigned short* feat16 = (unsigned short*)carve(p, (size_t)NODES*CC*2);
  unsigned short* Wq16 = (unsigned short*)carve(p, (size_t)CC*CC*2);
  unsigned short* Wk16 = (unsigned short*)carve(p, (size_t)CC*CC*2);
  unsigned short* Wv16 = (unsigned short*)carve(p, (size_t)CC*CC*2);
  unsigned short* Wp16 = (unsigned short*)carve(p, (size_t)CC*CC*2);
  unsigned short* W116 = (unsigned short*)carve(p, (size_t)FF*CC*2);
  unsigned short* W216 = (unsigned short*)carve(p, (size_t)CC*FF*2);
  float* qf    = (float*)carve(p, (size_t)NODES*CC*4);     // dead after edge pass 3
  float* kf    = (float*)carve(p, (size_t)NODES*CC*4);     // dead after edge pass 3
  float* vf    = (float*)carve(p, (size_t)NODES*CC*4);     // dead after edge pass 3
  float* score = (float*)carve(p, (size_t)EE*HH*4);        // dead after edge pass 3
  unsigned int* smaxe = (unsigned int*)carve(p, (size_t)NODES*HH*4);
  float* den   = (float*)carve(p, (size_t)NODES*HH*4);
  float* wv    = (float*)carve(p, (size_t)NODES*CC*4);
  float* xres  = (float*)carve(p, (size_t)NODES*CC*4);
  // aliases onto dead regions:
  unsigned short* h116 = (unsigned short*)qf;     // NODES*FF*2 == 102.4MB == qf+kf
  unsigned short* ln16 = (unsigned short*)vf;     // 25.6MB <= 51.2MB
  unsigned short* wv16 = (unsigned short*)score;  // 25.6MB <= 51.2MB

  const dim3 B(256);

  // ---- stage bf16 copies ----
  cvt_f32_bf16_kernel<<<ceil_div((size_t)NODES*CC,256), B, 0, stream>>>(feat, feat16, (size_t)NODES*CC);
  cvt_f32_bf16_kernel<<<ceil_div((size_t)CC*CC,256), B, 0, stream>>>(Wq, Wq16, (size_t)CC*CC);
  cvt_f32_bf16_kernel<<<ceil_div((size_t)CC*CC,256), B, 0, stream>>>(Wk, Wk16, (size_t)CC*CC);
  cvt_f32_bf16_kernel<<<ceil_div((size_t)CC*CC,256), B, 0, stream>>>(Wv, Wv16, (size_t)CC*CC);
  cvt_f32_bf16_kernel<<<ceil_div((size_t)CC*CC,256), B, 0, stream>>>(Wp, Wp16, (size_t)CC*CC);
  cvt_f32_bf16_kernel<<<ceil_div((size_t)FF*CC,256), B, 0, stream>>>(W1, W116, (size_t)FF*CC);
  cvt_f32_bf16_kernel<<<ceil_div((size_t)CC*FF,256), B, 0, stream>>>(W2, W216, (size_t)CC*FF);

  // ---- QKV projections (bf16 WMMA, f32 accumulate; B strip via LDS) ----
  launch_gemm(feat16, Wq16, bq, nullptr, qf, nullptr, NODES, CC, CC, 0, stream);
  launch_gemm(feat16, Wk16, bk, nullptr, kf, nullptr, NODES, CC, CC, 0, stream);
  launch_gemm(feat16, Wv16, bv, nullptr, vf, nullptr, NODES, CC, CC, 0, stream);

  // ---- init accumulators (ws is poisoned; must re-init every call) ----
  fill_u32_kernel<<<ceil_div((size_t)NODES*HH,256), B, 0, stream>>>(smaxe, 0x0u, (size_t)NODES*HH);
  fill_f32_kernel<<<ceil_div((size_t)NODES*HH,256), B, 0, stream>>>(den, 0.0f, (size_t)NODES*HH);
  fill_f32_kernel<<<ceil_div((size_t)NODES*CC,256), B, 0, stream>>>(wv, 0.0f, (size_t)NODES*CC);

  // ---- edge attention: score -> segment softmax -> weighted scatter ----
  edge_score_kernel<<<ceil_div((size_t)EE,8), B, 0, stream>>>(esrc, edst, qf, kf, score, smaxe);
  edge_exp_kernel<<<ceil_div((size_t)EE*HH,256), B, 0, stream>>>(edst, score, smaxe, den);
  edge_aggr_kernel<<<ceil_div((size_t)EE,8), B, 0, stream>>>(esrc, edst, score, den, vf, wv);

  // ---- output projection + residual: x = feat + wv @ Wp^T + bp ----
  cvt_f32_bf16_kernel<<<ceil_div((size_t)NODES*CC,256), B, 0, stream>>>(wv, wv16, (size_t)NODES*CC);
  launch_gemm(wv16, Wp16, bp, feat, xres, nullptr, NODES, CC, CC, 0, stream);

  // ---- layernorm -> bf16 ----
  ln_kernel<<<ceil_div((size_t)NODES,8), B, 0, stream>>>(xres, lng, lnb, ln16);

  // ---- FFN: gelu(ln @ W1^T + b1) @ W2^T + b2, residual x ----
  launch_gemm(ln16, W116, b1, nullptr, nullptr, h116, NODES, FF, CC, 1, stream);
  launch_gemm(h116, W216, b2, xres, out, nullptr, NODES, CC, FF, 0, stream);
}
